// MultiHeadDistanceLayer_57921928954261
// MI455X (gfx1250) — matrix operand order repaired and checked by reading, not compiled, not run on hardware
//
#include <hip/hip_runtime.h>
#include <hip/hip_bf16.h>

typedef __attribute__((ext_vector_type(16))) _Float16 v16h;
typedef __attribute__((ext_vector_type(8)))  float    v8f;

#define Bn  2
#define Ln  2048
#define Dn  128
#define Hn  8
#define HDn 32
#define NC  256   // H*HD

// Gather WMMA A-operand (16x32 f16) from an LDS tile with given row stride.
// Lane l (l<16): row M=l, K in {0..15}; lane l+16: row M=l, K in {16..31}.
__device__ inline v16h gather_a(const _Float16* base, int rowStride, int kb, int lane) {
  int row = lane & 15, hi = lane >> 4;
  const _Float16* p0 = base + row * rowStride + kb + 16 * hi;
  const _Float16* p1 = p0 + 8;
  v16h a;
#pragma unroll
  for (int i = 0; i < 8; ++i) { a[i] = p0[i]; a[8 + i] = p1[i]; }
  return a;
}

// ---------------- Kernel 0: convert Wq/Wk (128x256 f32) -> f16 transposed [256][128]
__global__ __launch_bounds__(256) void wconv_kernel(const float* __restrict__ Wq,
                                                    const float* __restrict__ Wk,
                                                    _Float16* __restrict__ WqT,
                                                    _Float16* __restrict__ WkT) {
  int idx = blockIdx.x * 256 + threadIdx.x;      // 2 * 256 * 128 = 65536
  int which = idx >> 15;
  int r   = idx & 32767;
  int col = r >> 7;                              // 0..255
  int k   = r & 127;                             // 0..127
  const float* W = which ? Wk : Wq;
  _Float16*    T = which ? WkT : WqT;
  T[(size_t)col * Dn + k] = (_Float16)W[(size_t)k * NC + col];
}

// ---------------- Kernel 1: QKV projections (WMMA) + reversed sigmoid gate
__global__ __launch_bounds__(256) void qkv_kernel(const float* __restrict__ x,
                                                  const float* __restrict__ pe,
                                                  const float* __restrict__ bq,
                                                  const float* __restrict__ bk,
                                                  const float* __restrict__ Wv,
                                                  const _Float16* __restrict__ WqT,
                                                  const _Float16* __restrict__ WkT,
                                                  _Float16* __restrict__ Qh,
                                                  _Float16* __restrict__ Kh,
                                                  float* __restrict__ vrev) {
  __shared__ _Float16 ax[16][128];
  int rt = blockIdx.x, tid = threadIdx.x;
  int r0 = rt * 16;
  int b  = r0 / Ln;
  int l0 = r0 % Ln;

  // Load xp = x + pe tile as f16 into LDS
  for (int i = tid; i < 16 * 128; i += 256) {
    int rr = i >> 7, cc = i & 127;
    int l = l0 + rr;
    float v = x[((size_t)b * Ln + l) * Dn + cc] + pe[(size_t)l * Dn + cc];
    ax[rr][cc] = (_Float16)v;
  }
  __syncthreads();

  // V = sigmoid(x @ Wv), stored reversed along key axis: vrev[h][b][L-1-l]
  if (tid < 128) {
    int rr = tid >> 3, h = tid & 7;
    int l = l0 + rr;
    const float* xr = x + ((size_t)b * Ln + l) * Dn;
    float s = 0.f;
#pragma unroll 4
    for (int c = 0; c < Dn; ++c) s += xr[c] * Wv[c * Hn + h];
    float sig = 1.f / (1.f + __expf(-s));
    vrev[((size_t)h * Bn + b) * Ln + (Ln - 1 - l)] = sig;
  }

  int w = tid >> 5, lane = tid & 31;
  int col = lane & 15, hi = lane >> 4;

  for (int which = 0; which < 2; ++which) {
    const _Float16* WT  = which ? WkT : WqT;
    const float*    bia = which ? bk  : bq;
    _Float16*       Out = which ? Kh  : Qh;
    for (int sub = 0; sub < 2; ++sub) {
      int n0 = w * 32 + 16 * sub;
      v8f c = {};
#pragma unroll
      for (int ks = 0; ks < 4; ++ks) {
        int kb = ks * 32;
        v16h a  = gather_a(&ax[0][0], 128, kb, lane);
        v16h bm = *(const v16h*)(WT + (size_t)(n0 + col) * Dn + kb + 16 * hi);
        c = __builtin_amdgcn_wmma_f32_16x16x32_f16(false, a, false, bm, (short)0, c, false, false);
      }
      float bv = bia[n0 + col];
#pragma unroll
      for (int r = 0; r < 8; ++r) {
        int l = l0 + r + 8 * hi;
        Out[((size_t)b * Ln + l) * NC + n0 + col] = (_Float16)(c[r] + bv);
      }
    }
  }
}

// ---------------- Kernel 2: attention + diagonal-weighted reduction into S
#define SCS 2052   // padded row stride for the score tile (16-byte aligned, bank-friendly)

__global__ __launch_bounds__(256) void attn_kernel(const _Float16* __restrict__ Qh,
                                                   const _Float16* __restrict__ Kh,
                                                   const float* __restrict__ vrev,
                                                   float* __restrict__ S) {
  extern __shared__ char smem[];
  float*    sc     = (float*)smem;            // 16 * SCS
  float*    Slocal = sc + 16 * SCS;           // 2048
  float*    rowmax = Slocal + 2048;           // 16
  float*    rowzi  = rowmax + 16;             // 16
  _Float16* qa     = (_Float16*)(rowzi + 16); // 16 * 32

  int bx = blockIdx.x;
  int qt = bx & 127;           // L/16 = 128 q-tiles
  int b  = (bx >> 7) & 1;
  int h  = bx >> 8;
  int q0 = qt * 16;
  int tid = threadIdx.x;
  int w = tid >> 5, lane = tid & 31;
  int col = lane & 15, hi = lane >> 4;

  for (int i = tid; i < 512; i += 256) {
    int rr = i >> 5, cc = i & 31;
    qa[i] = Qh[((size_t)b * Ln + q0 + rr) * NC + h * HDn + cc];
  }
  for (int i = tid; i < 2048; i += 256) Slocal[i] = 0.f;
  __syncthreads();

  // Scores: Q_tile (16x32) x K^T (32x2048), one WMMA per 16-key tile
  v16h a = gather_a(qa, 32, 0, lane);
  const float scale = 0.17677669529663687f;   // 1/sqrt(32)
  for (int i = 0; i < 16; ++i) {
    int kt  = w + 8 * i;
    int n0  = kt * 16;
    int key = n0 + col;
    v16h bm = *(const v16h*)(Kh + ((size_t)b * Ln + key) * NC + h * HDn + 16 * hi);
    v8f c = {};
    c = __builtin_amdgcn_wmma_f32_16x16x32_f16(false, a, false, bm, (short)0, c, false, false);
#pragma unroll
    for (int r = 0; r < 8; ++r)
      sc[(r + 8 * hi) * SCS + n0 + col] = c[r] * scale;
  }
  __syncthreads();

  // Per-row softmax stats (each wave owns 2 rows)
  for (int rr = 0; rr < 2; ++rr) {
    int row = 2 * w + rr;
    const float* sr = sc + row * SCS;
    float mx = -1e30f;
    for (int cix = lane; cix < 2048; cix += 32) mx = fmaxf(mx, sr[cix]);
#pragma unroll
    for (int off = 16; off; off >>= 1) mx = fmaxf(mx, __shfl_xor(mx, off, 32));
    float sm = 0.f;
    for (int cix = lane; cix < 2048; cix += 32) sm += __expf(sr[cix] - mx);
#pragma unroll
    for (int off = 16; off; off >>= 1) sm += __shfl_xor(sm, off, 32);
    if (lane == 0) { rowmax[row] = mx; rowzi[row] = 1.f / sm; }
  }
  __syncthreads();

  // Diagonal accumulation: S[m] += softmax(q,k) * vrev[k], m = k - q
  float vv[8];
#pragma unroll
  for (int j = 0; j < 8; ++j) vv[j] = vrev[((size_t)h * Bn + b) * Ln + tid + 256 * j];
  for (int q = 0; q < 16; ++q) {
    float mx = rowmax[q], zi = rowzi[q];
    const float* sr = sc + q * SCS;
    int qg = q0 + q;
#pragma unroll
    for (int j = 0; j < 8; ++j) {
      int k = tid + 256 * j;
      int m = k - qg;
      if (m >= 0) {
        float p = __expf(sr[k] - mx) * zi * vv[j];
        atomicAdd(&Slocal[m], p);           // ds_add_f32
      }
    }
  }
  __syncthreads();

  float* Sg = S + ((size_t)h * Bn + b) * Ln;
  for (int i = tid; i < 2048; i += 256)
    unsafeAtomicAdd(&Sg[i], Slocal[i]);     // global_atomic_add_f32
}

// ---------------- Kernel 3: 3-tap SAME average pool (divisor excludes padding)
__global__ __launch_bounds__(256) void pool_kernel(const float* __restrict__ S,
                                                   float* __restrict__ out) {
  int idx = blockIdx.x * 256 + threadIdx.x;   // B*L*H = 32768
  if (idx >= Bn * Ln * Hn) return;
  int h = idx & 7;
  int m = (idx >> 3) & (Ln - 1);
  int b = idx >> 14;
  const float* Sg = S + ((size_t)h * Bn + b) * Ln;
  float acc = Sg[m];
  int cnt = 1;
  if (m > 0)      { acc += Sg[m - 1]; cnt++; }
  if (m < Ln - 1) { acc += Sg[m + 1]; cnt++; }
  out[idx] = acc / (float)cnt;
}

extern "C" void kernel_launch(void* const* d_in, const int* in_sizes, int n_in,
                              void* d_out, int out_size, void* d_ws, size_t ws_size,
                              hipStream_t stream) {
  const float* x  = (const float*)d_in[0];
  const float* pe = (const float*)d_in[1];
  const float* Wq = (const float*)d_in[2];
  const float* bq = (const float*)d_in[3];
  const float* Wk = (const float*)d_in[4];
  const float* bk = (const float*)d_in[5];
  const float* Wv = (const float*)d_in[6];
  float* out = (float*)d_out;

  char* ws = (char*)d_ws;
  _Float16* WqT  = (_Float16*)(ws + 0);            //  64 KB
  _Float16* WkT  = (_Float16*)(ws + 65536);        //  64 KB
  _Float16* Qh   = (_Float16*)(ws + 131072);       //   2 MB
  _Float16* Kh   = (_Float16*)(ws + 2228224);      //   2 MB
  float*    vrev = (float*)   (ws + 4325376);      // 128 KB
  float*    S    = (float*)   (ws + 4456448);      // 128 KB

  (void)in_sizes; (void)n_in; (void)out_size; (void)ws_size;

  hipMemsetAsync(S, 0, (size_t)Hn * Bn * Ln * sizeof(float), stream);

  wconv_kernel<<<256, 256, 0, stream>>>(Wq, Wk, WqT, WkT);
  qkv_kernel<<<(Bn * Ln) / 16, 256, 0, stream>>>(x, pe, bq, bk, Wv, WqT, WkT, Qh, Kh, vrev);

  const size_t smem = (size_t)(16 * SCS + 2048 + 32) * sizeof(float) + 512 * sizeof(_Float16);
  (void)hipFuncSetAttribute((const void*)attn_kernel,
                            hipFuncAttributeMaxDynamicSharedMemorySize, (int)smem);
  attn_kernel<<<Hn * Bn * (Ln / 16), 256, smem, stream>>>(Qh, Kh, vrev, S);

  pool_kernel<<<(Bn * Ln * Hn + 255) / 256, 256, 0, stream>>>(S, out);
}
